// BaselineMultiStepRNN_53566832116434
// MI455X (gfx1250) — matrix operand order repaired and checked by reading, not compiled, not run on hardware
//
#include <hip/hip_runtime.h>
#include <hip/hip_bf16.h>

// ---------------------------------------------------------------------------
// Fused multi-step RNN for MI455X (gfx1250, wave32, WMMA).
//   h_{t+1} = tanh([x_t | cap_t] @ W_ih^T + h_t @ W_hh^T + b_ih + b_hh)
//   drop    = h_{t+1} @ fc_w^T + fc_b ;  cap_{t+1} = cap_t - drop ; out = cap
// One block owns a 16-row batch tile for all T steps; weights (320x256 f16)
// are held in registers as WMMA B-fragments for the whole recurrence.
// ---------------------------------------------------------------------------

#define T_STEPS 512
#define B_SZ    2048
#define F_DIM   63
#define IN_DIM  64
#define H_DIM   256
#define K_TOT   320      // IN_DIM + H_DIM
#define KT      10       // K_TOT / 32 k-tiles per WMMA chain
#define MT      16       // batch rows per block
#define NTW     2        // N-tiles (16 cols) per wave; 8 waves * 2 = 16 tiles = 256 cols

typedef _Float16 f16;
typedef __attribute__((ext_vector_type(16))) _Float16 v16h;
typedef __attribute__((ext_vector_type(8)))  float    v8f;

// Scheduling hints: force [DS-read / WMMA] interleave for double buffering.
#if __has_builtin(__builtin_amdgcn_sched_group_barrier)
#define SGB(mask, n) __builtin_amdgcn_sched_group_barrier((mask), (n), 0)
#else
#define SGB(mask, n)
#endif

// Branch-free tanh: prefer gfx1250 hardware v_tanh_f32; otherwise use
// exp2+rcp (saturates correctly to +/-1, no EXEC divergence).
__device__ __forceinline__ float fast_tanh(float x) {
#if __has_builtin(__builtin_amdgcn_tanhf)
    return __builtin_amdgcn_tanhf(x);
#else
    // tanh(x) = 1 - 2/(exp(2x)+1);  2*log2(e) = 2.8853900817779268
    float e = __builtin_amdgcn_exp2f(x * 2.8853900817779268f);
    return 1.0f - 2.0f * __builtin_amdgcn_rcpf(e + 1.0f);
#endif
}

// ---------------------------------------------------------------------------
// Prep: pack [W_ih^T ; W_hh^T] as f16 WcT[n][k] (row-major [256][320]) so the
// main kernel's per-lane B-fragment loads are contiguous 16B global loads.
// Also fuse the two bias vectors.
// ---------------------------------------------------------------------------
__global__ void rnn_prep_kernel(const float* __restrict__ Wih,
                                const float* __restrict__ Wihb,
                                const float* __restrict__ Whh,
                                const float* __restrict__ Whhb,
                                f16* __restrict__ WcT,
                                float* __restrict__ bias) {
    const int n = blockIdx.x;      // 0..255  (output neuron)
    const int k = threadIdx.x;     // 0..319  (input feature of concat space)
    float v = (k < IN_DIM) ? Wih[n * IN_DIM + k] : Whh[n * H_DIM + (k - IN_DIM)];
    WcT[(size_t)n * K_TOT + k] = (f16)v;
    if (k == 0) bias[n] = Wihb[n] + Whhb[n];
}

// ---------------------------------------------------------------------------
// Main recurrence kernel.
// ---------------------------------------------------------------------------
__launch_bounds__(256, 1)
__global__ void rnn_scan_kernel(const float* __restrict__ x,      // [B,T,F]
                                const float* __restrict__ seed,   // [B,1]
                                const f16*   __restrict__ WcT,    // [256][320] f16
                                const float* __restrict__ bias,   // [256]
                                const float* __restrict__ fcw,    // [256]
                                const float* __restrict__ fcb,    // [1]
                                float* __restrict__ out)          // [B,T]
{
    // LDS tiles (row padded by 8 halfwords -> 16B-aligned rows, 4-bank row skew)
    __shared__ __align__(16) f16 inpS[MT][IN_DIM + 8];   // [16][72]  K = 0..63
    __shared__ __align__(16) f16 hS  [MT][H_DIM + 8];    // [16][264] K = 64..319
    __shared__ float capS[MT];
    __shared__ float dropS[MT];

    const int tid   = threadIdx.x;
    const int lane  = tid & 31;
    const int wave  = tid >> 5;          // 0..7
    const int l15   = lane & 15;
    const int half  = lane >> 4;         // 0 or 1 (lane half for WMMA layouts)
    const int bBase = blockIdx.x * MT;

    // ---- init LDS state ----------------------------------------------------
    for (int i = tid; i < MT * (H_DIM + 8); i += 256) ((f16*)hS)[i] = (f16)0.f;
    if (tid < MT) { capS[tid] = seed[bBase + tid]; dropS[tid] = 0.f; }
    for (int i = tid; i < MT * IN_DIM; i += 256) {
        int r = i >> 6, f = i & 63;
        if (f < F_DIM)
            inpS[r][f] = (f16)x[(size_t)(bBase + r) * T_STEPS * F_DIM + f];
    }
    if (tid < MT) inpS[tid][F_DIM] = (f16)capS[tid];

    // ---- loop-invariant per-lane loads: W B-fragments, bias, fc weight -----
    // B layout (16-bit, 32x16): lane holds column n = l15; elems 0..7 -> K = half*8 + e, elems 8..15 -> K = 16 + half*8 + (e-8).
    v16h bfrag[NTW][KT];
    float biasv[NTW], fcv[NTW];
#pragma unroll
    for (int j = 0; j < NTW; ++j) {
        const int ncol = (wave * NTW + j) * 16 + l15;
        biasv[j] = bias[ncol];
        fcv[j]   = fcw[ncol];
#pragma unroll
        for (int kt = 0; kt < KT; ++kt) {
            const f16* p = WcT + (size_t)ncol * K_TOT + kt * 32 + half * 8;
            union { v16h v; uint4 u[2]; } tm;
            tm.u[0] = *(const uint4*)(p);
            tm.u[1] = *(const uint4*)(p + 16);
            bfrag[j][kt] = tm.v;
        }
    }
    const float fcbv = fcb[0];

    // Loop-invariant A-fragment LDS pointers.
    // A layout (16-bit, 16x32): lane row = l15;
    //   elems 0..7  -> K = kt*32 + half*8 + (0..7)
    //   elems 8..15 -> K = kt*32 + 16 + half*8 + (0..7)
    const f16* aptr[KT];
#pragma unroll
    for (int kt = 0; kt < KT; ++kt)
        aptr[kt] = (kt < 2) ? &inpS[l15][kt * 32 + half * 8]
                            : &hS  [l15][(kt - 2) * 32 + half * 8];

    __syncthreads();

    // ---- recurrence --------------------------------------------------------
    for (int t = 0; t < T_STEPS; ++t) {
        // accumulators start at the fused bias (C layout: col = l15, rows in regs)
        v8f acc[NTW];
#pragma unroll
        for (int j = 0; j < NTW; ++j)
#pragma unroll
            for (int i = 0; i < 8; ++i) acc[j][i] = biasv[j];

        // Software-pipelined K-chain: one-tile lookahead so the next tile's
        // ds_load_b128 pair is in flight while the current tile's WMMAs run.
        union AV { v16h v; uint4 u[2]; };
        AV ta[2];
        ta[0].u[0] = *(const uint4*)(aptr[0]);
        ta[0].u[1] = *(const uint4*)(aptr[0] + 16);
#pragma unroll
        for (int kt = 0; kt < KT; ++kt) {
            const int cur = kt & 1;
            if (kt + 1 < KT) {
                ta[cur ^ 1].u[0] = *(const uint4*)(aptr[kt + 1]);
                ta[cur ^ 1].u[1] = *(const uint4*)(aptr[kt + 1] + 16);
            }
            const v16h a = ta[cur].v;
#pragma unroll
            for (int j = 0; j < NTW; ++j)
                acc[j] = __builtin_amdgcn_wmma_f32_16x16x32_f16(
                    false, a, false, bfrag[j][kt], (short)0, acc[j], false, false);
        }
        // Enforce the pipelined schedule:
        //   DS(4) [tiles 0,1] ; 8x { WMMA(2) ; DS(2) } ; WMMA(4)
        // so loads for tile kt+1 issue before tile kt's WMMAs (double buffer).
        SGB(0x100, 4);                 // 4 ds reads (tiles 0 and 1)
#pragma unroll
        for (int kt = 0; kt < KT - 2; ++kt) {
            SGB(0x008, 2);             // 2 WMMAs of tile kt
            SGB(0x100, 2);             // 2 ds reads of tile kt+2
        }
        SGB(0x008, 4);                 // last two tiles' WMMAs

        // nonlinearity (branch-free; hardware v_tanh_f32 when available)
        float hn[NTW][8];
#pragma unroll
        for (int j = 0; j < NTW; ++j)
#pragma unroll
            for (int i = 0; i < 8; ++i) hn[j][i] = fast_tanh(acc[j][i]);

        __syncthreads();   // all WMMA reads of inpS/hS finished

        // write h_{t+1}, reduce drop = h_{t+1} . fc_w per row
#pragma unroll
        for (int j = 0; j < NTW; ++j) {
            const int ncol = (wave * NTW + j) * 16 + l15;
            float part[8];
#pragma unroll
            for (int i = 0; i < 8; ++i) {
                hS[i + half * 8][ncol] = (f16)hn[j][i];
                part[i] = hn[j][i] * fcv[j];
            }
            // butterfly across the 16-lane column group (stays within a half)
#pragma unroll
            for (int m = 1; m < 16; m <<= 1)
#pragma unroll
                for (int i = 0; i < 8; ++i) part[i] += __shfl_xor(part[i], m, 32);
            if (l15 == 0) {
#pragma unroll
                for (int i = 0; i < 8; ++i)
                    atomicAdd(&dropS[i + half * 8], part[i]);
            }
        }

        // stage x_{t+1} (and prefetch x_{t+2}) while the reduction settles
        if (t + 1 < T_STEPS) {
            const size_t tb = (size_t)(t + 1) * F_DIM;
            for (int i = tid; i < MT * IN_DIM; i += 256) {
                int r = i >> 6, f = i & 63;
                if (f < F_DIM)
                    inpS[r][f] = (f16)x[(size_t)(bBase + r) * T_STEPS * F_DIM + tb + f];
            }
            if (t + 2 < T_STEPS && tid < MT)
                __builtin_prefetch(
                    &x[(size_t)(bBase + tid) * T_STEPS * F_DIM + (size_t)(t + 2) * F_DIM],
                    0, 0);
        }

        __syncthreads();   // dropS complete, hS/inpS written

        if (tid < MT) {
            const float cap = capS[tid] - (dropS[tid] + fcbv);
            capS[tid]  = cap;
            dropS[tid] = 0.f;
            out[(size_t)(bBase + tid) * T_STEPS + t] = cap;
            inpS[tid][F_DIM] = (f16)cap;   // capacity column of next inp
        }

        __syncthreads();   // cap/drop updates visible before next K-chain
    }
}

// ---------------------------------------------------------------------------
// Launch. Workspace layout: [0,163840) WcT f16[256][320]; then bias f32[256].
// ---------------------------------------------------------------------------
extern "C" void kernel_launch(void* const* d_in, const int* in_sizes, int n_in,
                              void* d_out, int out_size, void* d_ws, size_t ws_size,
                              hipStream_t stream) {
    const float* x    = (const float*)d_in[0];   // [B,T,F]
    const float* seed = (const float*)d_in[1];   // [B,1]
    const float* Wih  = (const float*)d_in[2];   // [256,64]
    const float* Wihb = (const float*)d_in[3];   // [256]
    const float* Whh  = (const float*)d_in[4];   // [256,256]
    const float* Whhb = (const float*)d_in[5];   // [256]
    const float* fcw  = (const float*)d_in[6];   // [1,256]
    const float* fcb  = (const float*)d_in[7];   // [1]
    // d_in[8] = forecast_steps (== T_STEPS, fixed by the reference shapes)

    f16*   WcT  = (f16*)d_ws;
    float* bias = (float*)((char*)d_ws + (size_t)H_DIM * K_TOT * sizeof(f16));

    rnn_prep_kernel<<<H_DIM, K_TOT, 0, stream>>>(Wih, Wihb, Whh, Whhb, WcT, bias);
    rnn_scan_kernel<<<B_SZ / MT, 256, 0, stream>>>(
        x, seed, WcT, bias, fcw, fcb, (float*)d_out);
}
